// GCN_12403865551290
// MI455X (gfx1250) — compile-verified, matrix operand order checked by reading, not run on hardware
//
#include <hip/hip_runtime.h>

#define NNODES 100000
#define NFEATS 512
#define NHID   64
#define NCLS   40
#define NEDGES 1600000

typedef __attribute__((ext_vector_type(16))) __bf16 v16bf;
typedef __attribute__((ext_vector_type(8)))  float  v8f;
typedef __attribute__((ext_vector_type(4)))  float  fvec4;

union Frag16 { unsigned int u[8]; v16bf v; };

__device__ __forceinline__ unsigned int pack2_bf16(float a, float b) {
#if __has_builtin(__builtin_amdgcn_cvt_pk_bf16_f32)
  auto r = __builtin_amdgcn_cvt_pk_bf16_f32(a, b);        // v_cvt_pk_bf16_f32
  return __builtin_bit_cast(unsigned int, r);
#else
  unsigned int ua = __float_as_uint(a) + 0x8000u;          // round-half-up
  unsigned int ub = __float_as_uint(b) + 0x8000u;
#if __has_builtin(__builtin_amdgcn_perm)
  return __builtin_amdgcn_perm(ub, ua, 0x07060302u);       // v_perm_b32 -> {b.hi16, a.hi16}
#else
  return (ua >> 16) | (ub & 0xFFFF0000u);
#endif
#endif
}

// ---------------- degree / normalization ----------------
__global__ void k_fill1(float* p, int n) {
  int i = blockIdx.x * blockDim.x + threadIdx.x;
  if (i < n) p[i] = 1.0f;                 // self-loop contributes 1 to degree
}
__global__ void k_count(const int* __restrict__ dst, float* __restrict__ deg) {
  int e = blockIdx.x * blockDim.x + threadIdx.x;
  if (e < NEDGES) atomicAdd(&deg[dst[e]], 1.0f);
}
__global__ void k_rsqrt(float* p, int n) {
  int i = blockIdx.x * blockDim.x + threadIdx.x;
  if (i < n) p[i] = rsqrtf(p[i]);
}

// ---------------- GEMM1: h = x @ W1^T  (bf16 WMMA) ----------------
// grid: 3125 blocks x 32 rows, block: 256 threads = 8 waves.
// wave w -> (mtile = w>>2, ntile = w&3); K staged 64 wide -> 2 WMMAs / barrier pair.
__global__ __launch_bounds__(256) void gcn_gemm1(const float* __restrict__ x,
                                                 const float* __restrict__ W1,
                                                 float* __restrict__ h) {
  __shared__ unsigned int sA[32 * 32];   // 32 rows x 64 bf16 (32 u32 per row)
  __shared__ unsigned int sB[64 * 32];   // 64 n-rows x 64 bf16 (n-major)
  const int t     = threadIdx.x;
  const int wave  = t >> 5, lane = t & 31;
  const int half  = lane >> 4, mn = lane & 15;
  const int mtile = wave >> 2, ntile = wave & 3;
  const long row0 = (long)blockIdx.x * 32;

  // staging coordinates (fixed across K loop)
  const int am = t >> 3, ac = (t & 7) << 3;     // A: row 0..31, col group of 8
  const int bn = t >> 2, bc = (t & 3) << 4;     // B: n-row 0..63, col group of 16

  v8f acc = {};
  for (int kk = 0; kk < NFEATS; kk += 64) {
    // stage A tile: x[row0..row0+31][kk..kk+63] -> bf16 (non-temporal: single-use stream)
    {
      const float* ap = x + (row0 + am) * NFEATS + kk + ac;
      const fvec4 f0 = __builtin_nontemporal_load((const fvec4*)ap);
      const fvec4 f1 = __builtin_nontemporal_load((const fvec4*)(ap + 4));
      sA[am * 32 + (ac >> 1) + 0] = pack2_bf16(f0[0], f0[1]);
      sA[am * 32 + (ac >> 1) + 1] = pack2_bf16(f0[2], f0[3]);
      sA[am * 32 + (ac >> 1) + 2] = pack2_bf16(f1[0], f1[1]);
      sA[am * 32 + (ac >> 1) + 3] = pack2_bf16(f1[2], f1[3]);
      if (kk + 64 < NFEATS) __builtin_prefetch(ap + 64, 0, 0);  // global_prefetch_b8
    }
    // stage B tile: B[k][n] = W1[n][kk+k], stored n-major (row n holds k=0..63)
    {
      const float* wp = W1 + bn * NFEATS + kk + bc;
#pragma unroll
      for (int j = 0; j < 4; ++j) {
        const fvec4 f = *(const fvec4*)(wp + 4 * j);
        sB[bn * 32 + (bc >> 1) + 2 * j]     = pack2_bf16(f[0], f[1]);
        sB[bn * 32 + (bc >> 1) + 2 * j + 1] = pack2_bf16(f[2], f[3]);
      }
    }
    __syncthreads();
#pragma unroll
    for (int ks = 0; ks < 2; ++ks) {           // two K=32 slabs
      Frag16 a, b;
#pragma unroll
      for (int v = 0; v < 8; ++v) {
        const int ki = ks * 16 + (v >> 2) * 8 + half * 4 + (v & 3);  // u32 K-pair idx
        a.u[v] = sA[(mtile * 16 + mn) * 32 + ki];
        b.u[v] = sB[(ntile * 16 + mn) * 32 + ki];
      }
      acc = __builtin_amdgcn_wmma_f32_16x16x32_bf16(false, a.v, false, b.v,
                                                    (short)0, acc, false, false);
    }
    __syncthreads();
  }
#pragma unroll
  for (int r = 0; r < 8; ++r) {
    const int m = mtile * 16 + r + 8 * half;
    const int n = ntile * 16 + mn;
    h[(row0 + m) * NHID + n] = acc[r];
  }
}

// ---------------- layer-1 aggregation ----------------
__global__ void k_init1(const float* __restrict__ h, const float* __restrict__ dinv,
                        const float* __restrict__ b1, float* __restrict__ out1) {
  int idx = blockIdx.x * blockDim.x + threadIdx.x;
  if (idx < NNODES * NHID) {
    int i = idx >> 6, f = idx & 63;
    out1[idx] = b1[f] + dinv[i] * dinv[i] * h[idx];   // bias + self-loop
  }
}
__global__ void k_scat1(const int* __restrict__ src, const int* __restrict__ dst,
                        const float* __restrict__ dinv, const float* __restrict__ h,
                        float* __restrict__ out1) {
  int t = blockIdx.x * blockDim.x + threadIdx.x;
  if (t < NEDGES * 16) {
    int e = t >> 4, f0 = (t & 15) << 2;
    int s = src[e], d = dst[e];
    float nm = dinv[s] * dinv[d];
    const float* hp = h + (long)s * NHID + f0;
    float* op = out1 + (long)d * NHID + f0;
    atomicAdd(op + 0, nm * hp[0]);
    atomicAdd(op + 1, nm * hp[1]);
    atomicAdd(op + 2, nm * hp[2]);
    atomicAdd(op + 3, nm * hp[3]);
  }
}

// ---------------- GEMM2: z = relu(out1) @ W2^T (bf16 WMMA, N padded 40->48) ----------------
// grid: 3125 blocks x 32 rows, block 192 threads = 6 waves = 2 mtiles x 3 ntiles.
// Single K pass (K=64 -> 2 chained WMMAs), one barrier per block.
__global__ __launch_bounds__(192) void gcn_gemm2(const float* __restrict__ h1,
                                                 const float* __restrict__ W2,
                                                 float* __restrict__ z) {
  __shared__ unsigned int sA[32 * 32];   // 32 rows x 64 bf16
  __shared__ unsigned int sB[48 * 32];   // 48 n-rows x 64 bf16 (rows 40..47 zero)
  const int t     = threadIdx.x;
  const int wave  = t >> 5, lane = t & 31;
  const int half  = lane >> 4, mn = lane & 15;
  const int mtile = wave / 3, ntile = wave % 3;
  const long row0 = (long)blockIdx.x * 32;

  for (int idx = t; idx < 32 * 32; idx += 192) {       // A tile with fused ReLU
    const int m = idx >> 5, kc = (idx & 31) << 1;
    float f0 = h1[(row0 + m) * NHID + kc];
    float f1 = h1[(row0 + m) * NHID + kc + 1];
    f0 = f0 > 0.f ? f0 : 0.f;
    f1 = f1 > 0.f ? f1 : 0.f;
    sA[idx] = pack2_bf16(f0, f1);
  }
  for (int idx = t; idx < 48 * 32; idx += 192) {       // B tile, zero-padded n
    const int n = idx >> 5, kc = (idx & 31) << 1;
    float f0 = 0.f, f1 = 0.f;
    if (n < NCLS) {
      f0 = W2[n * NHID + kc];
      f1 = W2[n * NHID + kc + 1];
    }
    sB[idx] = pack2_bf16(f0, f1);
  }
  __syncthreads();

  v8f acc = {};
#pragma unroll
  for (int ks = 0; ks < 2; ++ks) {
    Frag16 a, b;
#pragma unroll
    for (int v = 0; v < 8; ++v) {
      const int ki = ks * 16 + (v >> 2) * 8 + half * 4 + (v & 3);
      a.u[v] = sA[(mtile * 16 + mn) * 32 + ki];
      b.u[v] = sB[(ntile * 16 + mn) * 32 + ki];
    }
    acc = __builtin_amdgcn_wmma_f32_16x16x32_bf16(false, a.v, false, b.v,
                                                  (short)0, acc, false, false);
  }
#pragma unroll
  for (int r = 0; r < 8; ++r) {
    const int m = mtile * 16 + r + 8 * half;
    const int n = ntile * 16 + mn;
    if (n < NCLS) z[(row0 + m) * NCLS + n] = acc[r];
  }
}

// ---------------- layer-2 aggregation ----------------
__global__ void k_init2(const float* __restrict__ z, const float* __restrict__ dinv,
                        const float* __restrict__ b2, float* __restrict__ out) {
  int idx = blockIdx.x * blockDim.x + threadIdx.x;
  if (idx < NNODES * NCLS) {
    int i = idx / NCLS, f = idx % NCLS;
    out[idx] = b2[f] + dinv[i] * dinv[i] * z[idx];
  }
}
__global__ void k_scat2(const int* __restrict__ src, const int* __restrict__ dst,
                        const float* __restrict__ dinv, const float* __restrict__ z,
                        float* __restrict__ out) {
  int t = blockIdx.x * blockDim.x + threadIdx.x;
  if (t < NEDGES * 10) {
    int e = t / 10, f0 = (t % 10) << 2;
    int s = src[e], d = dst[e];
    float nm = dinv[s] * dinv[d];
    const float* zp = z + (long)s * NCLS + f0;
    float* op = out + (long)d * NCLS + f0;
    atomicAdd(op + 0, nm * zp[0]);
    atomicAdd(op + 1, nm * zp[1]);
    atomicAdd(op + 2, nm * zp[2]);
    atomicAdd(op + 3, nm * zp[3]);
  }
}

extern "C" void kernel_launch(void* const* d_in, const int* in_sizes, int n_in,
                              void* d_out, int out_size, void* d_ws, size_t ws_size,
                              hipStream_t stream) {
  const float* x  = (const float*)d_in[0];
  const int*   ei = (const int*)d_in[1];     // int32 (JAX x64-off)
  const float* W1 = (const float*)d_in[2];
  const float* b1 = (const float*)d_in[3];
  const float* W2 = (const float*)d_in[4];
  const float* b2 = (const float*)d_in[5];
  float* out = (float*)d_out;

  char* ws = (char*)d_ws;
  float* dinv = (float*)ws;                                        // 400 KB
  float* h    = (float*)(ws + (size_t)(1u << 19));                 // 25.6 MB
  float* out1 = (float*)(ws + (size_t)(1u << 19) + 26u * 1048576); // 25.6 MB
  float* z    = (float*)(ws + (size_t)(1u << 19) + 52u * 1048576); // 16.0 MB

  const int* src = ei;
  const int* dst = ei + NEDGES;

  k_fill1<<<(NNODES + 255) / 256, 256, 0, stream>>>(dinv, NNODES);
  k_count<<<(NEDGES + 255) / 256, 256, 0, stream>>>(dst, dinv);
  k_rsqrt<<<(NNODES + 255) / 256, 256, 0, stream>>>(dinv, NNODES);

  gcn_gemm1<<<NNODES / 32, 256, 0, stream>>>(x, W1, h);

  k_init1<<<(NNODES * NHID + 255) / 256, 256, 0, stream>>>(h, dinv, b1, out1);
  k_scat1<<<(NEDGES * 16 + 255) / 256, 256, 0, stream>>>(src, dst, dinv, h, out1);

  gcn_gemm2<<<NNODES / 32, 192, 0, stream>>>(out1, W2, z);

  k_init2<<<(NNODES * NCLS + 255) / 256, 256, 0, stream>>>(z, dinv, b2, out);
  k_scat2<<<(NEDGES * 10 + 255) / 256, 256, 0, stream>>>(src, dst, dinv, z, out);
}
